// DeepSeekV3Block_16630113370893
// MI455X (gfx1250) — matrix operand order repaired and checked by reading, not compiled, not run on hardware
//
#include <hip/hip_runtime.h>
#include <hip/hip_bf16.h>
#include <math.h>

// ---------------------------------------------------------------------------
// Problem constants (from reference)
// ---------------------------------------------------------------------------
#define B_  2
#define S_  2048
#define D_  1024
#define H_  16
#define HD_ 64
#define R_  256
#define E_  8
#define MH_ 256
#define T_  (B_ * S_)     // 4096 tokens
#define EPS_ 1e-6f

typedef __attribute__((ext_vector_type(16))) __bf16 v16bf;
typedef __attribute__((ext_vector_type(8)))  float  v8f;

// fp32 -> bf16, round-to-nearest-even. Plain fptrunc lets the backend select
// the native packed v_cvt_*bf16* ops on gfx1250 instead of a scalar bit-twiddle
// chain.
__device__ __forceinline__ __bf16 f2bf(float f) { return (__bf16)f; }

// A fragment (16x32 bf16, ISA 7.12.2 layout):
//   elems 0..7  -> K = kbase + half*8 + i
//   elems 8..15 -> K = kbase + 16 + half*8 + (i-8)
__device__ __forceinline__ v16bf load_a_frag(const float* __restrict__ arow,
                                             int kbase, int half) {
    const float4* p0 = (const float4*)(arow + kbase + half * 8);
    const float4* p1 = (const float4*)(arow + kbase + 16 + half * 8);
    float4 a0 = p0[0], a1 = p0[1];
    float4 b0 = p1[0], b1 = p1[1];
    v16bf af;
    af[0]=f2bf(a0.x); af[1]=f2bf(a0.y); af[2]=f2bf(a0.z); af[3]=f2bf(a0.w);
    af[4]=f2bf(a1.x); af[5]=f2bf(a1.y); af[6]=f2bf(a1.z); af[7]=f2bf(a1.w);
    af[8]=f2bf(b0.x); af[9]=f2bf(b0.y); af[10]=f2bf(b0.z); af[11]=f2bf(b0.w);
    af[12]=f2bf(b1.x); af[13]=f2bf(b1.y); af[14]=f2bf(b1.z); af[15]=f2bf(b1.w);
    return af;
}

// B fragment (32x16 bf16): elem i -> K = kbase + half*8 + i (i<8), kbase + 16 + half*8 + (i-8) (i>=8), fixed column.
// Lane->column is the fast axis, so each of the 16 loads is wave-coalesced.
__device__ __forceinline__ v16bf load_b_frag(const float* __restrict__ bbase,
                                             int N) {
    v16bf bf;
    #pragma unroll
    for (int i = 0; i < 16; ++i)
        bf[i] = f2bf(bbase[(size_t)(i < 8 ? i : i + 8) * N]);
    return bf;
}

// ---------------------------------------------------------------------------
// GEMM: C[M,N] = op(A[M,K] @ Bm[K,N])
//   RS : multiply product by rowScale[row*rsStride]  (MoE gates)
//   RES: add residual[idx]                           (attention residual)
//   ACC: C += val instead of C = val                 (MoE accumulation)
// 256 threads = 8 waves; wave (wy,wx) computes a 16x16 tile; block = 32x64.
// K unrolled by 64 -> two v_wmma_f32_16x16x32_bf16 per iteration.
// ---------------------------------------------------------------------------
template <bool RS, bool RES, bool ACC>
__global__ __launch_bounds__(256)
void gemm_wmma_bf16(const float* __restrict__ A, const float* __restrict__ Bm,
                    float* __restrict__ C, int M, int N, int K,
                    const float* __restrict__ residual,
                    const float* __restrict__ rowScale, int rsStride) {
    const int lane = threadIdx.x & 31;
    const int wave = threadIdx.x >> 5;
    const int wy = wave >> 2;            // 0..1
    const int wx = wave & 3;             // 0..3
    const int tileM = blockIdx.y * 2 + wy;
    const int tileN = blockIdx.x * 4 + wx;
    const int half = lane >> 4;          // 0: lanes 0-15, 1: lanes 16-31
    const int r = lane & 15;

    const int rowA = tileM * 16 + r;     // A row this lane feeds
    const int col  = tileN * 16 + r;     // B/C column this lane feeds

    v8f acc = {};
    const float* arow  = A + (size_t)rowA * K;
    const float* bbase = Bm + (size_t)(half * 8) * N + col;
    const size_t bstep = (size_t)32 * N;

    for (int k0 = 0; k0 < K; k0 += 64) {
        if (k0 + 64 < K) {
            __builtin_prefetch(arow + k0 + 64, 0, 0);
            __builtin_prefetch(bbase + (size_t)(k0 + 64) * N, 0, 0);
        }
        v16bf a0f = load_a_frag(arow, k0, half);
        v16bf b0f = load_b_frag(bbase + (size_t)k0 * N, N);
        acc = __builtin_amdgcn_wmma_f32_16x16x32_bf16(
            false, a0f, false, b0f, (short)0, acc, false, false);

        v16bf a1f = load_a_frag(arow, k0 + 32, half);
        v16bf b1f = load_b_frag(bbase + (size_t)k0 * N + bstep, N);
        acc = __builtin_amdgcn_wmma_f32_16x16x32_bf16(
            false, a1f, false, b1f, (short)0, acc, false, false);
    }

    // C VGPR j -> row tileM*16 + half*8 + j, column = col (ISA 7.12.2)
    const int rbase = tileM * 16 + half * 8;
    #pragma unroll
    for (int j = 0; j < 8; ++j) {
        const int row = rbase + j;
        const size_t idx = (size_t)row * N + col;
        float val = acc[j];
        if constexpr (RS)  val *= rowScale[(size_t)row * rsStride];
        if constexpr (RES) val += residual[idx];
        if constexpr (ACC) C[idx] += val;
        else               C[idx] = val;
    }
}

// ---------------------------------------------------------------------------
// RMSNorm: one 256-thread block per row of D_ elements
// ---------------------------------------------------------------------------
__global__ __launch_bounds__(256)
void rmsnorm_kernel(const float* __restrict__ x, const float* __restrict__ w,
                    float* __restrict__ out) {
    __shared__ float red[256];
    const int row = blockIdx.x;
    const int tid = threadIdx.x;
    const float* xr = x + (size_t)row * D_;
    float ss = 0.f;
    for (int d = tid; d < D_; d += 256) { float v = xr[d]; ss += v * v; }
    red[tid] = ss;
    __syncthreads();
    for (int s = 128; s > 0; s >>= 1) {
        if (tid < s) red[tid] += red[tid + s];
        __syncthreads();
    }
    const float inv = rsqrtf(red[0] / (float)D_ + EPS_);
    float* orow = out + (size_t)row * D_;
    for (int d = tid; d < D_; d += 256) orow[d] = xr[d] * inv * w[d];
}

// ---------------------------------------------------------------------------
// RoPE in-place on q and k: one thread per (b,s,h,j), j < HD/2
// ---------------------------------------------------------------------------
__global__ __launch_bounds__(256)
void rope_kernel(float* __restrict__ q, float* __restrict__ k,
                 const int* __restrict__ pos_ids) {
    const int idx = blockIdx.x * blockDim.x + threadIdx.x;
    const int half = HD_ / 2;                       // 32
    if (idx >= B_ * S_ * H_ * half) return;
    const int j = idx & (half - 1);
    const int h = (idx / half) & (H_ - 1);
    const int s = (idx / (half * H_)) & (S_ - 1);
    const int b = idx / (half * H_ * S_);

    const float pos = (float)pos_ids[b * S_ + s];
    // inv = THETA^(-j/half) = exp(-j * ln(10000)/32)
    const float inv = __expf(-(float)j * (9.21034037197618f / (float)half));
    const float ang = pos * inv;
    const float cs = __cosf(ang), sn = __sinf(ang);

    const size_t base = ((size_t)(b * S_ + s) * H_ + h) * HD_;
    float q1 = q[base + j], q2 = q[base + half + j];
    q[base + j]        = q1 * cs - q2 * sn;
    q[base + half + j] = q1 * sn + q2 * cs;
    float k1 = k[base + j], k2 = k[base + half + j];
    k[base + j]        = k1 * cs - k2 * sn;
    k[base + half + j] = k1 * sn + k2 * cs;
}

// ---------------------------------------------------------------------------
// Causal attention, online softmax. One wave32 per (b,h,q); each lane owns
// head dims {lane, lane+32}. Wave-wide dot via shfl_xor butterfly.
// ---------------------------------------------------------------------------
__global__ __launch_bounds__(256)
void attention_kernel(const float* __restrict__ Q, const float* __restrict__ Kv,
                      const float* __restrict__ V, float* __restrict__ O) {
    const int wid = (blockIdx.x * blockDim.x + threadIdx.x) >> 5;
    const int lane = threadIdx.x & 31;
    if (wid >= B_ * H_ * S_) return;
    const int qi = wid & (S_ - 1);
    const int h = (wid / S_) & (H_ - 1);
    const int b = wid / (S_ * H_);

    const size_t qbase = ((size_t)(b * S_ + qi) * H_ + h) * HD_;
    const float scale = 0.125f;                     // 1/sqrt(64)
    const float q0 = Q[qbase + lane] * scale;
    const float q1 = Q[qbase + 32 + lane] * scale;

    float m = -3.0e38f, l = 0.f, a0 = 0.f, a1 = 0.f;
    for (int kk = 0; kk <= qi; ++kk) {
        const size_t kb = ((size_t)(b * S_ + kk) * H_ + h) * HD_;
        float p = q0 * Kv[kb + lane] + q1 * Kv[kb + 32 + lane];
        #pragma unroll
        for (int off = 16; off > 0; off >>= 1) p += __shfl_xor(p, off, 32);
        const float mn = fmaxf(m, p);
        const float sc = __expf(m - mn);
        const float w  = __expf(p - mn);
        l  = l * sc + w;
        a0 = a0 * sc + w * V[kb + lane];
        a1 = a1 * sc + w * V[kb + 32 + lane];
        m = mn;
    }
    const float inv = 1.0f / l;
    O[qbase + lane]      = a0 * inv;
    O[qbase + 32 + lane] = a1 * inv;
}

// ---------------------------------------------------------------------------
// Router: one wave32 per token. scores = sigmoid(x @ wr); top-2 over
// (scores + bias); gates = normalized raw scores at selected experts.
// ---------------------------------------------------------------------------
__global__ __launch_bounds__(256)
void router_kernel(const float* __restrict__ xn, const float* __restrict__ wr,
                   const float* __restrict__ bias, float* __restrict__ gates) {
    const int t = (blockIdx.x * blockDim.x + threadIdx.x) >> 5;
    const int lane = threadIdx.x & 31;
    if (t >= T_) return;
    float acc[E_];
    #pragma unroll
    for (int e = 0; e < E_; ++e) acc[e] = 0.f;
    const float* xr = xn + (size_t)t * D_;
    for (int d = lane; d < D_; d += 32) {
        const float xv = xr[d];
        const float* wrow = wr + (size_t)d * E_;
        #pragma unroll
        for (int e = 0; e < E_; ++e) acc[e] += xv * wrow[e];
    }
    #pragma unroll
    for (int e = 0; e < E_; ++e)
        #pragma unroll
        for (int off = 16; off > 0; off >>= 1)
            acc[e] += __shfl_xor(acc[e], off, 32);

    if (lane == 0) {
        float s[E_], sb[E_];
        #pragma unroll
        for (int e = 0; e < E_; ++e) {
            s[e] = 1.0f / (1.0f + __expf(-acc[e]));
            sb[e] = s[e] + bias[e];
        }
        int i0 = 0;
        #pragma unroll
        for (int e = 1; e < E_; ++e) if (sb[e] > sb[i0]) i0 = e;
        int i1 = (i0 == 0) ? 1 : 0;
        #pragma unroll
        for (int e = 0; e < E_; ++e)
            if (e != i0 && sb[e] > sb[i1]) i1 = e;
        const float denom = s[i0] + s[i1] + 1e-9f;
        float* g = gates + (size_t)t * E_;
        #pragma unroll
        for (int e = 0; e < E_; ++e) g[e] = 0.f;
        g[i0] = s[i0] / denom;
        g[i1] = s[i1] / denom;
    }
}

// ---------------------------------------------------------------------------
// hc = silu(h1) * h3
// ---------------------------------------------------------------------------
__global__ __launch_bounds__(256)
void silu_mul_kernel(const float* __restrict__ h1, const float* __restrict__ h3,
                     float* __restrict__ hc, int n) {
    const int i = blockIdx.x * blockDim.x + threadIdx.x;
    if (i >= n) return;
    const float a = h1[i];
    hc[i] = (a / (1.0f + __expf(-a))) * h3[i];
}

__global__ __launch_bounds__(256)
void copy_kernel(const float* __restrict__ src, float* __restrict__ dst, int n) {
    const int i = blockIdx.x * blockDim.x + threadIdx.x;
    if (i < n) dst[i] = src[i];
}

// ---------------------------------------------------------------------------
// Host-side orchestration
// ---------------------------------------------------------------------------
extern "C" void kernel_launch(void* const* d_in, const int* in_sizes, int n_in,
                              void* d_out, int out_size, void* d_ws, size_t ws_size,
                              hipStream_t stream) {
    const float* x        = (const float*)d_in[0];
    const int*   pos_ids  = (const int*)  d_in[1];
    const float* norm1_w  = (const float*)d_in[2];
    const float* wq       = (const float*)d_in[3];
    const float* wdkv     = (const float*)d_in[4];
    const float* wuk      = (const float*)d_in[5];
    const float* wuv      = (const float*)d_in[6];
    const float* wo       = (const float*)d_in[7];
    const float* norm2_w  = (const float*)d_in[8];
    const float* wr       = (const float*)d_in[9];
    const float* rbias    = (const float*)d_in[10];
    const float* w1       = (const float*)d_in[11];
    const float* w3       = (const float*)d_in[12];
    const float* w2       = (const float*)d_in[13];
    float* out = (float*)d_out;

    // Workspace carving (fp32). Buffers reused across phases.
    float* ws = (float*)d_ws;
    float* xn    = ws;                       // T*1024  (norm1 out; reused as norm2 out)
    float* q     = xn    + (size_t)T_*1024;  // T*1024  (q; reused as x1 = x + attn@wo)
    float* cbuf  = q     + (size_t)T_*1024;  // T*256   (latent c; reused as MoE h1)
    float* k     = cbuf  + (size_t)T_*256;   // T*1024
    float* v     = k     + (size_t)T_*1024;  // T*1024
    float* attn  = v     + (size_t)T_*1024;  // T*1024
    float* gates = attn  + (size_t)T_*1024;  // T*8
    float* h3    = gates + (size_t)T_*8;     // T*256
    float* hc    = h3    + (size_t)T_*256;   // T*256

    const int HH = H_ * HD_;                 // 1024

    // 1. xn = rmsnorm(x, norm1_w)
    rmsnorm_kernel<<<T_, 256, 0, stream>>>(x, norm1_w, xn);

    // 2-5. projections (bf16 WMMA)
    gemm_wmma_bf16<false,false,false><<<dim3(HH/64, T_/32), 256, 0, stream>>>(xn, wq,   q,    T_, HH,  D_, nullptr, nullptr, 0);
    gemm_wmma_bf16<false,false,false><<<dim3(R_/64, T_/32), 256, 0, stream>>>(xn, wdkv, cbuf, T_, R_,  D_, nullptr, nullptr, 0);
    gemm_wmma_bf16<false,false,false><<<dim3(HH/64, T_/32), 256, 0, stream>>>(cbuf, wuk, k,   T_, HH,  R_, nullptr, nullptr, 0);
    gemm_wmma_bf16<false,false,false><<<dim3(HH/64, T_/32), 256, 0, stream>>>(cbuf, wuv, v,   T_, HH,  R_, nullptr, nullptr, 0);

    // 6. RoPE in-place on q, k
    {
        const int n = B_ * S_ * H_ * (HD_ / 2);
        rope_kernel<<<(n + 255) / 256, 256, 0, stream>>>(q, k, pos_ids);
    }

    // 7. causal attention -> attn
    {
        const int waves = B_ * H_ * S_;
        attention_kernel<<<waves / 8, 256, 0, stream>>>(q, k, v, attn);
    }

    // 8. x1 = x + attn @ wo      (written into q buffer; q/k/v now dead)
    float* x1 = q;
    gemm_wmma_bf16<false,true,false><<<dim3(D_/64, T_/32), 256, 0, stream>>>(attn, wo, x1, T_, D_, HH, x, nullptr, 0);

    // 9. xn2 = rmsnorm(x1, norm2_w)   (into xn buffer)
    rmsnorm_kernel<<<T_, 256, 0, stream>>>(x1, norm2_w, xn);

    // 10. router -> gates[T, E]
    router_kernel<<<(T_ * 32) / 256, 256, 0, stream>>>(xn, wr, rbias, gates);

    // 11. out = x1, then accumulate gated expert outputs
    copy_kernel<<<(T_ * D_ + 255) / 256, 256, 0, stream>>>(x1, out, T_ * D_);

    float* h1 = cbuf;
    for (int e = 0; e < E_; ++e) {
        const float* w1e = w1 + (size_t)e * D_ * MH_;
        const float* w3e = w3 + (size_t)e * D_ * MH_;
        const float* w2e = w2 + (size_t)e * MH_ * D_;
        gemm_wmma_bf16<false,false,false><<<dim3(MH_/64, T_/32), 256, 0, stream>>>(xn, w1e, h1, T_, MH_, D_, nullptr, nullptr, 0);
        gemm_wmma_bf16<false,false,false><<<dim3(MH_/64, T_/32), 256, 0, stream>>>(xn, w3e, h3, T_, MH_, D_, nullptr, nullptr, 0);
        silu_mul_kernel<<<(T_ * MH_ + 255) / 256, 256, 0, stream>>>(h1, h3, hc, T_ * MH_);
        // out += gate[t,e] * (hc @ w2[e])
        gemm_wmma_bf16<true,false,true><<<dim3(D_/64, T_/32), 256, 0, stream>>>(hc, w2e, out, T_, D_, MH_, nullptr, gates + e, E_);
    }
}
